// CausalSelfAttention_newsim_weight_68350109549023
// MI455X (gfx1250) — matrix operand order repaired and checked by reading, not compiled, not run on hardware
//
#include <hip/hip_runtime.h>

#define TSEQ 2048
#define CDIM 512
#define NHEAD 8
#define HD 64
#define BT 4096   // B*T

typedef __attribute__((ext_vector_type(16))) __bf16 v16bf;
typedef __attribute__((ext_vector_type(8)))  float  v8f;
typedef unsigned uint4v __attribute__((ext_vector_type(4)));
typedef int      int8v  __attribute__((ext_vector_type(8)));
typedef int      int4v  __attribute__((ext_vector_type(4)));

union ABits { unsigned u[8]; v16bf v; };

__device__ __forceinline__ unsigned short f2bf_bits(float f) {
  unsigned a = __builtin_bit_cast(unsigned, f);
  a = (a + 0x7FFFu + ((a >> 16) & 1u)) >> 16;   // round-to-nearest-even
  return (unsigned short)a;
}
__device__ __forceinline__ unsigned packf2(float lo, float hi) {
  return (unsigned)f2bf_bits(lo) | ((unsigned)f2bf_bits(hi) << 16);
}

__device__ __forceinline__ v8f wmma_bf16(v16bf a, v16bf b, v8f c) {
  return __builtin_amdgcn_wmma_f32_16x16x32_bf16(false, a, false, b, (short)0, c,
                                                 false, false);
}

// ---------------- f32 -> bf16 bulk convert (one pass, bandwidth bound) -------
__global__ void __launch_bounds__(256)
cvt_bf16_kernel(const float* __restrict__ src, unsigned short* __restrict__ dst, int n) {
  const int i = (blockIdx.x * 256 + threadIdx.x) * 8;
  if (i + 8 > n) return;
  const float4 f0 = *(const float4*)(src + i);
  const float4 f1 = *(const float4*)(src + i + 4);
  unsigned* d = (unsigned*)(dst + i);
  d[0] = packf2(f0.x, f0.y);
  d[1] = packf2(f0.z, f0.w);
  d[2] = packf2(f1.x, f1.y);
  d[3] = packf2(f1.z, f1.w);
}

// ------- 64x64 register-blocked bf16 GEMM block: A[M x 512] * B[512 x 512]^T -
// 4x4 WMMA tiles per wave: each A/B fragment is reused 4x -> 32 FLOP/B.
__device__ __forceinline__ void gemm64x64_bf16(const unsigned short* __restrict__ A,
                                               const unsigned short* __restrict__ B,
                                               int m0, int n0, v8f acc[4][4]) {
  const int lane = threadIdx.x & 31;
  const int ln = lane & 15, g = lane >> 4;
  const unsigned short* arow[4];
  const unsigned short* brow[4];
#pragma unroll
  for (int i = 0; i < 4; ++i) {
    arow[i] = A + (size_t)(m0 + 16*i + ln) * CDIM;
    brow[i] = B + (size_t)(n0 + 16*i + ln) * CDIM;
  }
  for (int kk = 0; kk < CDIM; kk += 32) {
    ABits af[4], bfr[4];
#pragma unroll
    for (int i = 0; i < 4; ++i)
#pragma unroll
      for (int p = 0; p < 8; ++p)
        af[i].u[p] = *(const unsigned*)(arow[i] + kk + 2*(p&3) + 16*(p>>2) + 8*g);
#pragma unroll
    for (int j = 0; j < 4; ++j)
#pragma unroll
      for (int p = 0; p < 8; ++p)
        bfr[j].u[p] = *(const unsigned*)(brow[j] + kk + 2*(p&3) + 16*(p>>2) + 8*g);
#pragma unroll
    for (int i = 0; i < 4; ++i)
#pragma unroll
      for (int j = 0; j < 4; ++j)
        acc[i][j] = wmma_bf16(af[i].v, bfr[j].v, acc[i][j]);
  }
}

// v0 = x @ Wv^T + bv -> v0bf (Q/K) and vwbf = v0*w[row] (V), both bf16
__global__ void __launch_bounds__(256)
proj_v_kernel(const unsigned short* __restrict__ xbf, const unsigned short* __restrict__ wvbf,
              const float* __restrict__ bv, const float* __restrict__ w,
              unsigned short* __restrict__ v0bf, unsigned short* __restrict__ vwbf) {
  const int wg = blockIdx.x * 8 + (threadIdx.x >> 5);
  const int n_t = wg & 7;              // 8 col blocks of 64
  const int m_t = wg >> 3;             // 64 row blocks of 64
  const int lane = threadIdx.x & 31, ln = lane & 15, g = lane >> 4;
  const int m0 = m_t * 64, n0 = n_t * 64;
  v8f acc[4][4];
  const v8f vzero = {0.f,0.f,0.f,0.f,0.f,0.f,0.f,0.f};
#pragma unroll
  for (int i = 0; i < 4; ++i)
#pragma unroll
    for (int j = 0; j < 4; ++j) acc[i][j] = vzero;
  gemm64x64_bf16(xbf, wvbf, m0, n0, acc);
#pragma unroll
  for (int j = 0; j < 4; ++j) {
    const int n = n0 + 16*j + ln;
    const float bias = bv[n];
#pragma unroll
    for (int i = 0; i < 4; ++i)
#pragma unroll
      for (int r = 0; r < 8; ++r) {
        const int m = m0 + 16*i + r + 8*g;
        const float v0 = acc[i][j][r] + bias;
        v0bf[(size_t)m * CDIM + n] = f2bf_bits(v0);
        vwbf[(size_t)m * CDIM + n] = f2bf_bits(v0 * w[m]);
      }
  }
}

// out = y @ Wp^T + bp (f32 output)
__global__ void __launch_bounds__(256)
proj_out_kernel(const unsigned short* __restrict__ ybf, const unsigned short* __restrict__ wpbf,
                const float* __restrict__ bp, float* __restrict__ out) {
  const int wg = blockIdx.x * 8 + (threadIdx.x >> 5);
  const int n_t = wg & 7;
  const int m_t = wg >> 3;
  const int lane = threadIdx.x & 31, ln = lane & 15, g = lane >> 4;
  const int m0 = m_t * 64, n0 = n_t * 64;
  v8f acc[4][4];
  const v8f vzero = {0.f,0.f,0.f,0.f,0.f,0.f,0.f,0.f};
#pragma unroll
  for (int i = 0; i < 4; ++i)
#pragma unroll
    for (int j = 0; j < 4; ++j) acc[i][j] = vzero;
  gemm64x64_bf16(ybf, wpbf, m0, n0, acc);
#pragma unroll
  for (int j = 0; j < 4; ++j) {
    const int n = n0 + 16*j + ln;
    const float bias = bp[n];
#pragma unroll
    for (int i = 0; i < 4; ++i)
#pragma unroll
      for (int r = 0; r < 8; ++r) {
        const int m = m0 + 16*i + r + 8*g;
        out[(size_t)m * CDIM + n] = acc[i][j][r] + bias;
      }
  }
}

// ---------------- TDM: async 32x64 bf16 tile (row pitch 512 elems) -> LDS ----
// LDS receives padded rows: 32 dwords data + 1 dword pad => pitch 66 shorts.
#define KV_PITCH 66
#define WAVE_LDS_SHORTS (2*32*KV_PITCH + 16*32)   // Ktile + Vtile + P = 4736

__device__ __forceinline__ void tdm_load_2d(unsigned lds_byte_addr,
                                            const unsigned short* gbase,
                                            unsigned long long elem_off) {
  const unsigned long long ga = (unsigned long long)gbase + elem_off * 2ull;
  uint4v g0;
  g0[0] = 1u;                                   // count=1, no gather
  g0[1] = lds_byte_addr;                        // lds_addr
  g0[2] = (unsigned)ga;                         // global_addr[31:0]
  g0[3] = (unsigned)(ga >> 32) | (2u << 30);    // global_addr[56:32] | type=2
  int8v g1;
  g1[0] = (int)((1u << 16) | (1u << 20) | (4u << 22)); // data_size=2B, pad_en, 32dw interval, 1dw pad
  g1[1] = (int)(64u << 16);                     // tensor_dim0 = 64
  g1[2] = (int)(32u << 16);                     // tensor_dim1 = 32
  g1[3] = (int)(64u << 16);                     // tile_dim0 = 64
  g1[4] = (int)(32u);                           // tile_dim1 = 32, tile_dim2 = 0
  g1[5] = (int)(512u);                          // tensor_dim0_stride = 512
  g1[6] = 0;
  g1[7] = 0;
  const int4v gz4 = {0, 0, 0, 0};
  const int8v gz8 = {0, 0, 0, 0, 0, 0, 0, 0};
  // 6-arg toolchain variant: (g0, g1, g2, g3, extra, cpol)
  __builtin_amdgcn_tensor_load_to_lds(g0, g1, gz4, gz4, gz8, 0);
}

// ---------------- flash attention: one wave = 16 query rows of one (b,h) -----
__global__ void __launch_bounds__(128)
flash_kernel(const unsigned short* __restrict__ v0bf,
             const unsigned short* __restrict__ vwbf,
             const float* __restrict__ w, unsigned short* __restrict__ ybf) {
  __shared__ unsigned short smem[4 * WAVE_LDS_SHORTS];
  const int wv = __builtin_amdgcn_readfirstlane((int)threadIdx.x) >> 5; // wave-uniform
  const int lane = threadIdx.x & 31, ln = lane & 15, g = lane >> 4;
  const int W = blockIdx.x * 4 + wv;
  const int qt = W & 127;
  const int h  = (W >> 7) & 7;
  const int b  = W >> 10;
  const int q0 = qt * 16;
  const size_t rowbase = (size_t)b * TSEQ;

  unsigned short* ldsK = smem + wv * WAVE_LDS_SHORTS;
  unsigned short* ldsV = ldsK + 32 * KV_PITCH;
  unsigned short* ldsP = ldsV + 32 * KV_PITCH;
  const unsigned ldsK_addr = (unsigned)(size_t)(void*)ldsK;  // low 32 bits = LDS offset
  const unsigned ldsV_addr = (unsigned)(size_t)(void*)ldsV;

  // Q: two 16x32 bf16 A-tiles resident in registers
  ABits aq[2];
  {
    const unsigned short* qrow = v0bf + (rowbase + q0 + ln) * CDIM + h * HD;
#pragma unroll
    for (int t = 0; t < 2; ++t)
#pragma unroll
      for (int p = 0; p < 8; ++p)
        aq[t].u[p] = *(const unsigned*)(qrow + 32*t + 2*(p&3) + 16*(p>>2) + 8*g);
  }

  float m_i[8], l_i[8];
  v8f acc[4];
  const v8f vzero = {0.f,0.f,0.f,0.f,0.f,0.f,0.f,0.f};
#pragma unroll
  for (int r = 0; r < 8; ++r) { m_i[r] = -3.0e38f; l_i[r] = 0.f; }
#pragma unroll
  for (int n = 0; n < 4; ++n) acc[n] = vzero;

  const int qmaxrow = q0 + 15;
  for (int s0 = 0; s0 <= qmaxrow; s0 += 32) {
    // previous iteration's ds_loads must retire before TDM overwrites the tiles
    asm volatile("s_wait_dscnt 0" ::: "memory");
    tdm_load_2d(ldsK_addr, v0bf, (unsigned long long)(rowbase + s0) * CDIM + h * HD);
    tdm_load_2d(ldsV_addr, vwbf, (unsigned long long)(rowbase + s0) * CDIM + h * HD);
    __builtin_amdgcn_s_wait_tensorcnt(0);

    float lg[2][8];
    // ---- S = Q K^T : K B-tiles straight out of LDS ----
#pragma unroll
    for (int c2 = 0; c2 < 2; ++c2) {
      ABits bk[2];
#pragma unroll
      for (int t = 0; t < 2; ++t)
#pragma unroll
        for (int r = 0; r < 8; ++r)
          bk[t].u[r] = *(const unsigned*)&ldsK[(16*c2 + ln) * KV_PITCH + 32*t + 2*(r&3) + 16*(r>>2) + 8*g];
      v8f s = vzero;
      s = wmma_bf16(aq[0].v, bk[0].v, s);
      s = wmma_bf16(aq[1].v, bk[1].v, s);
      const int keycol = s0 + 16 * c2 + ln;
      const float wk = w[rowbase + keycol] * 0.125f;     // w[s]/sqrt(64)
#pragma unroll
      for (int r = 0; r < 8; ++r) {
        const int qrow = q0 + r + 8 * g;
        const float v = s[r] * wk;
        lg[c2][r] = (keycol > qrow) ? -3.0e38f : v;
      }
    }
    // ---- online softmax (row stats across 16-lane groups) ----
    float alpha[8];
#pragma unroll
    for (int r = 0; r < 8; ++r) {
      float v = fmaxf(lg[0][r], lg[1][r]);
      v = fmaxf(v, __shfl_xor(v, 1));
      v = fmaxf(v, __shfl_xor(v, 2));
      v = fmaxf(v, __shfl_xor(v, 4));
      v = fmaxf(v, __shfl_xor(v, 8));
      const float mn = fmaxf(m_i[r], v);
      alpha[r] = __expf(m_i[r] - mn);
      m_i[r] = mn;
      const float p0 = __expf(lg[0][r] - mn);
      const float p1 = __expf(lg[1][r] - mn);
      lg[0][r] = p0; lg[1][r] = p1;
      float rs = p0 + p1;
      rs += __shfl_xor(rs, 1);
      rs += __shfl_xor(rs, 2);
      rs += __shfl_xor(rs, 4);
      rs += __shfl_xor(rs, 8);
      l_i[r] = l_i[r] * alpha[r] + rs;
    }
#pragma unroll
    for (int n = 0; n < 4; ++n)
#pragma unroll
      for (int r = 0; r < 8; ++r)
        acc[n][r] *= alpha[r];

    // ---- P: C-layout -> LDS -> A-layout (16x32 bf16) ----
#pragma unroll
    for (int c2 = 0; c2 < 2; ++c2)
#pragma unroll
      for (int r = 0; r < 8; ++r)
        ldsP[(r + 8*g) * 32 + 16*c2 + ln] = f2bf_bits(lg[c2][r]);
    asm volatile("s_wait_dscnt 0" ::: "memory");
    ABits pa;
#pragma unroll
    for (int p = 0; p < 8; ++p)
      pa.u[p] = *(const unsigned*)&ldsP[ln * 32 + 2*(p&3) + 16*(p>>2) + 8*g];

    // ---- acc += P (16x32) @ V (32x64), V B-tiles from LDS ----
#pragma unroll
    for (int n = 0; n < 4; ++n) {
      ABits vb;
      const int dd = 16 * n + ln;
#pragma unroll
      for (int r = 0; r < 8; ++r) {
        const int kl = 16 * (r >> 2) + 8 * g + 2 * (r & 3);
        const unsigned lo = ldsV[kl * KV_PITCH + dd];
        const unsigned hi = ldsV[(kl + 1) * KV_PITCH + dd];
        vb.u[r] = lo | (hi << 16);
      }
      acc[n] = wmma_bf16(pa.v, vb.v, acc[n]);
    }
  }

  // ---- epilogue: y = acc / l, stored bf16 for the output GEMM ----
#pragma unroll
  for (int n = 0; n < 4; ++n)
#pragma unroll
    for (int r = 0; r < 8; ++r) {
      const int qrow = q0 + r + 8 * g;
      ybf[(rowbase + qrow) * CDIM + h * HD + 16 * n + ln] = f2bf_bits(acc[n][r] / l_i[r]);
    }
}

extern "C" void kernel_launch(void* const* d_in, const int* in_sizes, int n_in,
                              void* d_out, int out_size, void* d_ws, size_t ws_size,
                              hipStream_t stream) {
  (void)in_sizes; (void)n_in; (void)out_size; (void)ws_size;
  const float* x      = (const float*)d_in[0];
  const float* weight = (const float*)d_in[1];   // [B*T]
  const float* Wv     = (const float*)d_in[2];
  const float* bv     = (const float*)d_in[3];
  const float* Wp     = (const float*)d_in[4];
  const float* bp     = (const float*)d_in[5];
  float* out = (float*)d_out;

  unsigned short* v0bf = (unsigned short*)d_ws;                 // 4 MB
  unsigned short* vwbf = v0bf + (size_t)BT * CDIM;              // 4 MB
  unsigned short* ybf  = vwbf + (size_t)BT * CDIM;              // 4 MB
  unsigned short* xbf  = ybf  + (size_t)BT * CDIM;              // 4 MB
  unsigned short* wvbf = xbf  + (size_t)BT * CDIM;              // 0.5 MB
  unsigned short* wpbf = wvbf + (size_t)CDIM * CDIM;            // 0.5 MB

  const dim3 blk256(256);
  const int nx = BT * CDIM, nw = CDIM * CDIM;
  cvt_bf16_kernel<<<nx / 8 / 256, blk256, 0, stream>>>(x, xbf, nx);
  cvt_bf16_kernel<<<nw / 8 / 256, blk256, 0, stream>>>(Wv, wvbf, nw);
  cvt_bf16_kernel<<<nw / 8 / 256, blk256, 0, stream>>>(Wp, wpbf, nw);

  const int gemm_blocks = (BT / 64) * (CDIM / 64) / 8;          // 64
  proj_v_kernel<<<gemm_blocks, blk256, 0, stream>>>(xbf, wvbf, bv, weight, v0bf, vwbf);

  const int flash_blocks = (2 * NHEAD * (TSEQ / 16)) / 4;       // 512
  flash_kernel<<<flash_blocks, dim3(128), 0, stream>>>(v0bf, vwbf, weight, ybf);

  proj_out_kernel<<<gemm_blocks, blk256, 0, stream>>>(ybf, wpbf, bp, out);
}
